// MultiHeadAttention_3599182594172
// MI455X (gfx1250) — matrix operand order repaired and checked by reading, not compiled, not run on hardware
//
#include <hip/hip_runtime.h>
#include <hip/hip_bf16.h>

// N=4096, D=1024, H=16, DK=64.  proj = X @ Wqkv; qk|v split; per-head
// softmax(Q Kt / 8) @ V with Q==K.  bf16 WMMA (16x16x32) + f32 accum.
// Attention K/V tiles staged LDS via TDM (tensor_load_to_lds), double-buffered.

#define NQ 4096
#define DMODEL 1024
#define HEADS 16
#define DKH 64

typedef __attribute__((ext_vector_type(16))) __bf16 v16bf;
typedef __attribute__((ext_vector_type(8)))  float  v8f;
typedef __attribute__((ext_vector_type(4)))  unsigned int u32x4;
typedef __attribute__((ext_vector_type(8)))  unsigned int u32x8;

union Frag {
    v16bf v;
    uint4 q[2];
};

static __device__ __forceinline__ v8f wmma_bf16(const Frag& a, const Frag& b, v8f c) {
    return __builtin_amdgcn_wmma_f32_16x16x32_bf16(
        /*neg_a=*/false, a.v, /*neg_b=*/false, b.v,
        /*c_mod=*/(short)0, c, /*reuse_a=*/false, /*reuse_b=*/false);
}

// ---------------------------------------------------------------------------
// TDM: DMA a 64x64 bf16 tile (row pitch `row_stride` elements) into LDS with
// 4-DWORD padding after each 128B row -> LDS row stride 72 elements (144 B).
// D# per CDNA5 ISA ch.8: group0 {count,lds,global,type=2}, group1 {dims}.
// ---------------------------------------------------------------------------
static __device__ __forceinline__ void tdm_load_tile64x64(unsigned lds_addr,
                                                          unsigned long long gaddr,
                                                          unsigned row_stride) {
    u32x4 g0;
    g0[0] = 1u;                                               // count=1, user D#
    g0[1] = lds_addr;                                         // lds_addr[31:0]
    g0[2] = (unsigned)gaddr;                                  // global_addr[31:0]
    g0[3] = (unsigned)((gaddr >> 32) & 0x01FFFFFFu)           // global_addr[56:32]
            | 0x80000000u;                                    // type=2 (image)
    u32x8 g1;
    g1[0] = (1u << 16)        // data_size = 2 bytes
          | (1u << 20)        // pad_enable
          | (4u << 22)        // pad_interval: 32 DWORDs (128 B row)
          | (3u << 25);       // pad_amount: 4 DWORDs (16 B) -> 144 B LDS pitch
    g1[1] = 64u << 16;        // tensor_dim0 = 64 elements
    g1[2] = 4096u << 16;      // tensor_dim1 = 4096 rows (OOB bound)
    g1[3] = 64u << 16;        // tile_dim0 = 64
    g1[4] = 64u;              // tile_dim1 = 64, tile_dim2 = 0
    g1[5] = row_stride;       // tensor_dim0_stride[31:0] (elements)
    g1[6] = 0u;
    g1[7] = 0u;
    asm volatile("tensor_load_to_lds %0, %1" :: "s"(g0), "s"(g1) : "memory");
}

// ---------------------------------------------------------------------------
// Prep: convert X (f32) -> Xb (bf16) row-major
// ---------------------------------------------------------------------------
__global__ __launch_bounds__(256) void cvt_x_kernel(const float* __restrict__ X,
                                                    __bf16* __restrict__ Xb, int n) {
    int i = blockIdx.x * 256 + threadIdx.x;
    if (i < n) Xb[i] = (__bf16)X[i];
}

// ---------------------------------------------------------------------------
// Prep: W [1024][2048] f32 -> Wt [2048][1024] bf16 (LDS tile transpose)
// ---------------------------------------------------------------------------
__global__ __launch_bounds__(256) void transpose_w_kernel(const float* __restrict__ W,
                                                          __bf16* __restrict__ Wt) {
    __shared__ __bf16 t[32][33];
    const int bx = blockIdx.x * 32;   // column (n) base, 0..2047
    const int by = blockIdx.y * 32;   // row (k) base,    0..1023
    const int x  = threadIdx.x;       // 0..31
    const int y0 = threadIdx.y;       // 0..7
#pragma unroll
    for (int i = 0; i < 4; i++) {
        int y = y0 + i * 8;
        t[y][x] = (__bf16)W[(size_t)(by + y) * (2 * DMODEL) + bx + x];
    }
    __syncthreads();
#pragma unroll
    for (int i = 0; i < 4; i++) {
        int y = y0 + i * 8;
        Wt[(size_t)(bx + y) * DMODEL + by + x] = t[x][y];
    }
}

// ---------------------------------------------------------------------------
// GEMM1: proj = Xb @ Wt^T  (i.e. X @ Wqkv).  64x64 block tile, 4 waves,
// each wave one 16x64 strip.  cols < 1024 -> qk (row major bf16),
// cols >= 1024 -> vt (transposed: vt[dk_global][n]).
// ---------------------------------------------------------------------------
__global__ __launch_bounds__(128) void qkv_kernel(const __bf16* __restrict__ Xb,
                                                  const __bf16* __restrict__ Wt,
                                                  __bf16* __restrict__ qk,
                                                  __bf16* __restrict__ vt) {
    const int lane = threadIdx.x & 31;
    const int wave = threadIdx.x >> 5;
    const int half = lane >> 4;   // 0: lanes 0-15, 1: lanes 16-31
    const int lcol = lane & 15;
    const int rb   = blockIdx.y * 64 + wave * 16;   // row base for this wave
    const int cb   = blockIdx.x * 64;               // column base (0..1984)

    v8f C[4] = {};
    const __bf16* xrow = Xb + (size_t)(rb + lcol) * DMODEL;

    for (int k = 0; k < DMODEL; k += 32) {
        Frag a;
        a.q[0] = *(const uint4*)(xrow + k + half * 16);
        a.q[1] = *(const uint4*)(xrow + k + half * 16 + 8);
#pragma unroll
        for (int nt = 0; nt < 4; nt++) {
            const __bf16* wrow = Wt + (size_t)(cb + nt * 16 + lcol) * DMODEL + k;
            Frag b;
            b.q[0] = *(const uint4*)(wrow + half * 16);
            b.q[1] = *(const uint4*)(wrow + half * 16 + 8);
            C[nt] = wmma_bf16(a, b, C[nt]);
        }
    }

    if (cb < DMODEL) {
        // qk region: row-major bf16 [N][D]
#pragma unroll
        for (int nt = 0; nt < 4; nt++) {
            int col = cb + nt * 16 + lcol;
#pragma unroll
            for (int r = 0; r < 8; r++)
                qk[(size_t)(rb + half * 8 + r) * DMODEL + col] = (__bf16)C[nt][r];
        }
    } else {
        // v region -> store transposed: vt[dk_global][n], 8 rows packed/lane
#pragma unroll
        for (int nt = 0; nt < 4; nt++) {
            int dkg = cb - DMODEL + nt * 16 + lcol;   // 0..1023
            union { __bf16 h[8]; uint4 q; } pack;
#pragma unroll
            for (int r = 0; r < 8; r++) pack.h[r] = (__bf16)C[nt][r];
            *(uint4*)(vt + (size_t)dkg * NQ + rb + half * 8) = pack.q;
        }
    }
}

// ---------------------------------------------------------------------------
// Flash attention: grid (N/64 row blocks, H heads), 4 waves x 32 lanes.
// K/V 64x64 tiles DMA'd into LDS by the TDM (wave 0), double-buffered and
// shared by all 4 waves; online softmax; P round-trips through LDS to become
// the A operand of the P@V WMMA.
// ---------------------------------------------------------------------------
__global__ __launch_bounds__(128) void attn_kernel(const __bf16* __restrict__ qk,
                                                   const __bf16* __restrict__ vt,
                                                   float* __restrict__ out) {
    __shared__ __align__(16) __bf16 Kt[2][64][72];   // TDM-padded K tiles
    __shared__ __align__(16) __bf16 Vt[2][64][72];   // TDM-padded V tiles
    __shared__ __align__(16) __bf16 Plds[4][16][72]; // wave-private P staging

    const int lane = threadIdx.x & 31;
    const int wave = threadIdx.x >> 5;
    const int half = lane >> 4;
    const int lcol = lane & 15;
    const int h    = blockIdx.y;
    const int qb   = blockIdx.x * 64 + wave * 16;

    const unsigned long long qk_base = (unsigned long long)(uintptr_t)qk;
    const unsigned long long vt_base = (unsigned long long)(uintptr_t)vt;

    // Q fragments (row M = lcol) from global, kept in registers throughout.
    const __bf16* qrow = qk + (size_t)(qb + lcol) * DMODEL + h * DKH;
    Frag aQ0, aQ1;
    aQ0.q[0] = *(const uint4*)(qrow + half * 16);
    aQ0.q[1] = *(const uint4*)(qrow + half * 16 + 8);
    aQ1.q[0] = *(const uint4*)(qrow + 32 + half * 16);
    aQ1.q[1] = *(const uint4*)(qrow + 32 + half * 16 + 8);

    // Prologue: DMA tiles for key block 0 into buffer 0.
    if (wave == 0) {
        tdm_load_tile64x64((unsigned)(uintptr_t)&Kt[0][0][0],
                           qk_base + 2ull * (size_t)(h * DKH), DMODEL);
        tdm_load_tile64x64((unsigned)(uintptr_t)&Vt[0][0][0],
                           vt_base + 2ull * ((size_t)h * DKH * NQ), NQ);
    }

    v8f   O[4] = {};
    float rowmax[8], rowsum[8];
#pragma unroll
    for (int r = 0; r < 8; r++) { rowmax[r] = -3.0e38f; rowsum[r] = 0.0f; }

    for (int kb = 0; kb < NQ / 64; kb++) {
        const int buf = kb & 1;

        if (wave == 0) __builtin_amdgcn_s_wait_tensorcnt(0);
        __syncthreads();   // tiles[buf] ready; everyone done reading buf^1

        if (wave == 0 && kb + 1 < NQ / 64) {
            const int ncb1 = (kb + 1) * 64;
            tdm_load_tile64x64((unsigned)(uintptr_t)&Kt[buf ^ 1][0][0],
                               qk_base + 2ull * ((size_t)ncb1 * DMODEL + h * DKH),
                               DMODEL);
            tdm_load_tile64x64((unsigned)(uintptr_t)&Vt[buf ^ 1][0][0],
                               vt_base + 2ull * ((size_t)h * DKH * NQ + ncb1),
                               NQ);
        }

        // ---- S = Q Kt for this wave's 16x64 strip ----
        v8f S[4];
#pragma unroll
        for (int nt = 0; nt < 4; nt++) {
            const __bf16* krow = &Kt[buf][nt * 16 + lcol][0];
            Frag b0, b1;
            b0.q[0] = *(const uint4*)(krow + half * 16);
            b0.q[1] = *(const uint4*)(krow + half * 16 + 8);
            b1.q[0] = *(const uint4*)(krow + 32 + half * 16);
            b1.q[1] = *(const uint4*)(krow + 32 + half * 16 + 8);
            v8f c = {};
            c = wmma_bf16(aQ0, b0, c);
            c = wmma_bf16(aQ1, b1, c);
            S[nt] = c;
        }

        // ---- online softmax (scale = 1/sqrt(64) = 0.125) ----
#pragma unroll
        for (int nt = 0; nt < 4; nt++)
#pragma unroll
            for (int r = 0; r < 8; r++) S[nt][r] *= 0.125f;

#pragma unroll
        for (int r = 0; r < 8; r++) {
            float m = fmaxf(fmaxf(S[0][r], S[1][r]), fmaxf(S[2][r], S[3][r]));
#pragma unroll
            for (int off = 1; off < 16; off <<= 1)
                m = fmaxf(m, __shfl_xor(m, off, 32));   // reduce within 16-lane half
            float mnew  = fmaxf(rowmax[r], m);
            float alpha = __expf(rowmax[r] - mnew);
            rowmax[r] = mnew;

            float psum = 0.0f;
#pragma unroll
            for (int nt = 0; nt < 4; nt++) {
                float p = __expf(S[nt][r] - mnew);
                Plds[wave][half * 8 + r][nt * 16 + lcol] = (__bf16)p;
                psum += p;
            }
            rowsum[r] = rowsum[r] * alpha + psum;
#pragma unroll
            for (int nt = 0; nt < 4; nt++) O[nt][r] *= alpha;
        }

        // wave-private LDS: order stores before the transposed A-frag reads
        asm volatile("s_wait_dscnt 0x0" ::: "memory");

        // ---- P A-fragments (row M = lcol) ----
        const __bf16* prow = &Plds[wave][lcol][0];
        Frag aP0, aP1;
        aP0.q[0] = *(const uint4*)(prow + half * 16);
        aP0.q[1] = *(const uint4*)(prow + half * 16 + 8);
        aP1.q[0] = *(const uint4*)(prow + 32 + half * 16);
        aP1.q[1] = *(const uint4*)(prow + 32 + half * 16 + 8);

        // ---- O += P @ V  (V B-frags from the TDM-staged LDS tile) ----
#pragma unroll
        for (int nt = 0; nt < 4; nt++) {
            const __bf16* vrow = &Vt[buf][nt * 16 + lcol][0];
            Frag b0, b1;
            b0.q[0] = *(const uint4*)(vrow + half * 16);
            b0.q[1] = *(const uint4*)(vrow + half * 16 + 8);
            b1.q[0] = *(const uint4*)(vrow + 32 + half * 16);
            b1.q[1] = *(const uint4*)(vrow + 32 + half * 16 + 8);
            O[nt] = wmma_bf16(aP0, b0, O[nt]);
            O[nt] = wmma_bf16(aP1, b1, O[nt]);
        }
    }

    // ---- normalize and write out[n][h*64 + dk] ----
#pragma unroll
    for (int r = 0; r < 8; r++) {
        float s = rowsum[r];
#pragma unroll
        for (int off = 1; off < 16; off <<= 1)
            s += __shfl_xor(s, off, 32);
        float inv = 1.0f / s;
        float* orow = out + (size_t)(qb + half * 8 + r) * DMODEL + h * DKH;
#pragma unroll
        for (int nt = 0; nt < 4; nt++)
            orow[nt * 16 + lcol] = O[nt][r] * inv;
    }
}

// ---------------------------------------------------------------------------
extern "C" void kernel_launch(void* const* d_in, const int* in_sizes, int n_in,
                              void* d_out, int out_size, void* d_ws, size_t ws_size,
                              hipStream_t stream) {
    const float* X = (const float*)d_in[0];   // [4096, 1024]
    const float* W = (const float*)d_in[1];   // [1024, 2048]
    float* out     = (float*)d_out;           // [4096, 1024]

    char* ws = (char*)d_ws;
    __bf16* qk = (__bf16*)(ws);                               //  8 MB [N][D]
    __bf16* vt = (__bf16*)(ws + (size_t)8  * 1024 * 1024);    //  8 MB [D][N]
    __bf16* Xb = (__bf16*)(ws + (size_t)16 * 1024 * 1024);    //  8 MB [N][D]
    __bf16* Wt = (__bf16*)(ws + (size_t)24 * 1024 * 1024);    //  4 MB [2D][D]

    cvt_x_kernel<<<(NQ * DMODEL + 255) / 256, 256, 0, stream>>>(X, Xb, NQ * DMODEL);
    transpose_w_kernel<<<dim3(2 * DMODEL / 32, DMODEL / 32), dim3(32, 8), 0, stream>>>(W, Wt);
    qkv_kernel<<<dim3(2 * DMODEL / 64, NQ / 64), 128, 0, stream>>>(Xb, Wt, qk, vt);
    attn_kernel<<<dim3(NQ / 64, HEADS), 128, 0, stream>>>(qk, vt, out);
}
